// TopKMoE_81200651698545
// MI455X (gfx1250) — matrix operand order repaired and checked by reading, not compiled, not run on hardware
//
#include <hip/hip_runtime.h>

#define B_ 16384
#define D_ 3072
#define H_ 128
#define O_ 10
#define E_ 8
#define NPAIR 28           // unordered top-2 expert pairs (8 choose 2)
#define MT 128             // rows per tile
#define TILES (B_ / MT)    // 128
#define NCHUNK (D_ / 32)   // 96

typedef _Float16 half_t;
typedef __attribute__((ext_vector_type(4)))  _Float16 v4h;
typedef __attribute__((ext_vector_type(8)))  _Float16 v8h;
typedef __attribute__((ext_vector_type(16))) _Float16 v16h;
typedef __attribute__((ext_vector_type(8)))  float    v8f;

union V16HU { v16h v; v8h h[2]; };

__device__ __forceinline__ v8f wmma16(v16h a, v16h b, v8f c) {
  return __builtin_amdgcn_wmma_f32_16x16x32_f16(false, a, false, b, (short)0, c,
                                                false, false);
}

__device__ __forceinline__ void vzero(v8f& a) {
#pragma unroll
  for (int i = 0; i < 8; ++i) a[i] = 0.0f;
}

// A-matrix 16x32 f16 fragment per ISA layout:
// lane<16: M=lane, K = kc+{0..7} then kc+{16..23}; lane>=16: K = kc+{8..15}, kc+{24..31}
__device__ __forceinline__ v16h ldsA(const half_t* A, int lda, int rowbase, int kc, int lane) {
  int l = lane & 15, hi = (lane >> 4) & 1;
  const half_t* p = A + (size_t)(rowbase + l) * lda + kc + hi * 8;
  V16HU f;
  f.h[0] = *(const v8h*)p;
  f.h[1] = *(const v8h*)(p + 16);
  return f.v;
}

// B-matrix 32x16 f16 fragment; Bt stored column-major in LDS: Bt[col][k]
// lane<16: N=lane, K = kc+{0..7} then kc+{16..23}; lane>=16: K = kc+{8..15}, kc+{24..31}
__device__ __forceinline__ v16h ldsB(const half_t* Bt, int ldb, int colbase, int kc, int lane) {
  int l = lane & 15, hi = (lane >> 4) & 1;
  const half_t* p = Bt + (size_t)(colbase + l) * ldb + kc + hi * 8;
  V16HU f;
  f.h[0] = *(const v8h*)p;
  f.h[1] = *(const v8h*)(p + 16);
  return f.v;
}

// ---------------------------------------------------------------------------
// Kernel 0: zero pair counters
// ---------------------------------------------------------------------------
__global__ void moe_zero(int* cnt) {
  if (threadIdx.x < NPAIR) cnt[threadIdx.x] = 0;
}

// ---------------------------------------------------------------------------
// Kernel 1: WMMA gate + top-2 softmax + pair-bucket scatter (pipelined)
// ---------------------------------------------------------------------------
__global__ __launch_bounds__(256) void moe_gate(
    const float* __restrict__ x, const float* __restrict__ gw, const float* __restrict__ gb,
    int* __restrict__ cnt, int* __restrict__ bucket,
    float* __restrict__ wlo, float* __restrict__ whi) {
  __shared__ half_t Bt[16][D_ + 8];   // gate_w transposed f16, cols 8..15 zero
  __shared__ half_t Ash[2][MT][40];   // double-buffered A tile (K=32, padded)
  __shared__ float  sc[MT][E_];
  __shared__ int    lcnt[NPAIR];
  __shared__ int    lbase[NPAIR];

  const int tid = threadIdx.x, lane = tid & 31, wid = tid >> 5;
  const int row0 = blockIdx.x * MT;

  for (int i = tid; i < 16 * D_; i += 256) {
    int d = i >> 4, e = i & 15;
    Bt[e][d] = (e < E_) ? (half_t)gw[d * E_ + e] : (half_t)0.f;
  }
  if (tid < NPAIR) lcnt[tid] = 0;

  const int r  = tid >> 1;
  const int hf = tid & 1;
  const float* xrow = x + (size_t)(row0 + r) * D_ + hf * 16;

  // prologue: stage chunk 0 into buffer 0
  {
    const float4* src = (const float4*)xrow;
    half_t* dst = &Ash[0][r][hf * 16];
#pragma unroll
    for (int j = 0; j < 4; ++j) {
      float4 v = src[j];
      v4h h = {(half_t)v.x, (half_t)v.y, (half_t)v.z, (half_t)v.w};
      *(v4h*)(dst + j * 4) = h;
    }
  }
  v8f acc; vzero(acc);
  __syncthreads();

  for (int ic = 0; ic < NCHUNK; ++ic) {
    const int kc = ic * 32;
    float4 pre[4];
    const bool more = (ic + 1) < NCHUNK;
    if (more) {  // issue next-chunk loads before compute (overlap)
      const float4* src = (const float4*)(xrow + kc + 32);
#pragma unroll
      for (int j = 0; j < 4; ++j) pre[j] = src[j];
    }
    v16h a = ldsA(&Ash[ic & 1][0][0], 40, wid * 16, 0, lane);
    v16h b = ldsB(&Bt[0][0], D_ + 8, 0, kc, lane);
    acc = wmma16(a, b, acc);
    if (more) {
      half_t* dst = &Ash[(ic + 1) & 1][r][hf * 16];
#pragma unroll
      for (int j = 0; j < 4; ++j) {
        v4h h = {(half_t)pre[j].x, (half_t)pre[j].y, (half_t)pre[j].z, (half_t)pre[j].w};
        *(v4h*)(dst + j * 4) = h;
      }
    }
    __syncthreads();
  }

  {  // scatter C tile (f32 16x16 layout) into score LDS
    int l = lane & 15, hi = lane >> 4;
    if (l < E_) {
#pragma unroll
      for (int i = 0; i < 8; ++i) sc[wid * 16 + (hi ? 8 + i : i)][l] = acc[i];
    }
  }
  __syncthreads();

  int pidx = 0, off = 0, packed = 0;
  float wl = 0.f, wh = 0.f;
  if (tid < MT) {
    float s[E_];
    float m = -1e30f; int a1 = 0;
#pragma unroll
    for (int e = 0; e < E_; ++e) {
      s[e] = sc[tid][e] + gb[e];
      if (s[e] > m) { m = s[e]; a1 = e; }
    }
    float m2 = -1e30f; int a2 = (a1 == 0) ? 1 : 0;
#pragma unroll
    for (int e = 0; e < E_; ++e)
      if (e != a1 && s[e] > m2) { m2 = s[e]; a2 = e; }
    float Z = 0.f;
#pragma unroll
    for (int e = 0; e < E_; ++e) Z += __expf(s[e] - m);
    float p1 = __expf(s[a1] - m) / Z;
    float p2 = __expf(s[a2] - m) / Z;
    float inv = 1.0f / (p1 + p2 + 1e-8f);
    float wtop = p1 * inv, wsec = p2 * inv;
    int elo = min(a1, a2), ehi = max(a1, a2);
    pidx = elo * (2 * E_ - elo - 1) / 2 + (ehi - elo - 1);
    wl = (elo == a1) ? wtop : wsec;
    wh = (ehi == a1) ? wtop : wsec;
    int slo = (elo == a1) ? 0 : 1;         // which output slot expert elo feeds
    packed = (row0 + tid) | (slo << 14);   // row fits in 14 bits
    off = atomicAdd(&lcnt[pidx], 1);
  }
  __syncthreads();
  if (tid < NPAIR) lbase[tid] = atomicAdd(&cnt[tid], lcnt[tid]);
  __syncthreads();
  if (tid < MT) {
    int idx = lbase[pidx] + off;
    bucket[pidx * B_ + idx] = packed;
    wlo[pidx * B_ + idx] = wl;
    whi[pidx * B_ + idx] = wh;
  }
}

// ---------------------------------------------------------------------------
// Kernel 2: per-pair gathered FFN (layer1 dual-expert pipelined, tail per expert)
// ---------------------------------------------------------------------------
struct SmemL1 {
  half_t Ash[2][MT][40];
  half_t B1lo[2][H_][40];
  half_t B1hi[2][H_][40];
};
struct SmemT {
  half_t W2t[H_][136];
  half_t W3t[16][136];
};
union SmemU { SmemL1 l1; SmemT t; };

__global__ __launch_bounds__(256) void moe_ffn(
    const float* __restrict__ x,
    const float* __restrict__ w1, const float* __restrict__ b1,
    const float* __restrict__ w2, const float* __restrict__ b2,
    const float* __restrict__ w3, const float* __restrict__ b3,
    const int* __restrict__ cnt, const int* __restrict__ bucket,
    const float* __restrict__ wlo, const float* __restrict__ whi,
    float* __restrict__ partial) {
  __shared__ SmemU u;
  __shared__ half_t Hbuf[MT][136];
  __shared__ half_t H2buf[MT][136];
  __shared__ float b1s[H_], b2s[H_], b3s[16];
  __shared__ int   rowp[MT];
  __shared__ float rwl[MT], rwh[MT];

  const int pidx = blockIdx.x / TILES;
  const int tb = (blockIdx.x % TILES) * MT;
  const int c = cnt[pidx];
  if (tb >= c) return;                 // block-uniform early exit
  const int nvalid = min(MT, c - tb);

  int ptmp = pidx, span = E_ - 1, el = 0;
  while (ptmp >= span) { ptmp -= span; ++el; --span; }
  const int eLo = el, eHi = el + 1 + ptmp;

  const int tid = threadIdx.x, lane = tid & 31, wid = tid >> 5;
  const float* w1lo = w1 + (size_t)eLo * D_ * H_;
  const float* w1hi = w1 + (size_t)eHi * D_ * H_;

  if (tid < MT) {
    int idx = tb + tid;
    if (idx < c) {
      rowp[tid] = bucket[pidx * B_ + idx];
      rwl[tid]  = wlo[pidx * B_ + idx];
      rwh[tid]  = whi[pidx * B_ + idx];
    } else {
      rowp[tid] = 0; rwl[tid] = 0.f; rwh[tid] = 0.f;
    }
  }
  __syncthreads();  // rowp visible before gathered prefetch

  const int ar = tid >> 1, ahf = tid & 1;     // A staging: row, k-half
  const int bn = tid & 127, bkg = tid >> 7;   // B staging: col, k-group

  v8f accL[8], accH[8];
#pragma unroll
  for (int i = 0; i < 8; ++i) { vzero(accL[i]); vzero(accH[i]); }

  // --- prologue: stage chunk 0 into buffer 0 ---
  {
    float4 a4[4]; float bl[16], bh[16];
    if (ar < nvalid) {
      int rid = rowp[ar] & 0x3FFF;
      const float4* src = (const float4*)(x + (size_t)rid * D_ + ahf * 16);
#pragma unroll
      for (int j = 0; j < 4; ++j) a4[j] = src[j];
    }
#pragma unroll
    for (int rep = 0; rep < 4; ++rep) {
      int k = bkg * 16 + rep * 4;
#pragma unroll
      for (int j = 0; j < 4; ++j) {
        bl[rep * 4 + j] = w1lo[(size_t)(k + j) * H_ + bn];
        bh[rep * 4 + j] = w1hi[(size_t)(k + j) * H_ + bn];
      }
    }
    if (ar < nvalid) {
      half_t* dst = &u.l1.Ash[0][ar][ahf * 16];
#pragma unroll
      for (int j = 0; j < 4; ++j) {
        v4h h = {(half_t)a4[j].x, (half_t)a4[j].y, (half_t)a4[j].z, (half_t)a4[j].w};
        *(v4h*)(dst + j * 4) = h;
      }
    }
#pragma unroll
    for (int rep = 0; rep < 4; ++rep) {
      int k = bkg * 16 + rep * 4;
      v4h hl = {(half_t)bl[rep * 4], (half_t)bl[rep * 4 + 1],
                (half_t)bl[rep * 4 + 2], (half_t)bl[rep * 4 + 3]};
      *(v4h*)&u.l1.B1lo[0][bn][k] = hl;
      v4h hh = {(half_t)bh[rep * 4], (half_t)bh[rep * 4 + 1],
                (half_t)bh[rep * 4 + 2], (half_t)bh[rep * 4 + 3]};
      *(v4h*)&u.l1.B1hi[0][bn][k] = hh;
    }
  }
  __syncthreads();

  // --- pipelined main loop: prefetch next chunk, compute current, one barrier ---
  for (int ic = 0; ic < NCHUNK; ++ic) {
    float4 a4[4]; float bl[16], bh[16];
    const bool more = (ic + 1) < NCHUNK;
    if (more) {
      const int kn = (ic + 1) * 32;
      if (ar < nvalid) {
        int rid = rowp[ar] & 0x3FFF;
        const float4* src = (const float4*)(x + (size_t)rid * D_ + kn + ahf * 16);
#pragma unroll
        for (int j = 0; j < 4; ++j) a4[j] = src[j];
      }
#pragma unroll
      for (int rep = 0; rep < 4; ++rep) {
        int k = bkg * 16 + rep * 4;
#pragma unroll
        for (int j = 0; j < 4; ++j) {
          bl[rep * 4 + j] = w1lo[(size_t)(kn + k + j) * H_ + bn];
          bh[rep * 4 + j] = w1hi[(size_t)(kn + k + j) * H_ + bn];
        }
      }
    }
    // compute current chunk from LDS
    {
      const int buf = ic & 1;
      v16h a = ldsA(&u.l1.Ash[buf][0][0], 40, wid * 16, 0, lane);
#pragma unroll
      for (int ct = 0; ct < 8; ++ct) {
        v16h bb = ldsB(&u.l1.B1lo[buf][0][0], 40, ct * 16, 0, lane);
        accL[ct] = wmma16(a, bb, accL[ct]);
      }
#pragma unroll
      for (int ct = 0; ct < 8; ++ct) {
        v16h bb = ldsB(&u.l1.B1hi[buf][0][0], 40, ct * 16, 0, lane);
        accH[ct] = wmma16(a, bb, accH[ct]);
      }
    }
    if (more) {
      const int nb = (ic + 1) & 1;
      if (ar < nvalid) {
        half_t* dst = &u.l1.Ash[nb][ar][ahf * 16];
#pragma unroll
        for (int j = 0; j < 4; ++j) {
          v4h h = {(half_t)a4[j].x, (half_t)a4[j].y, (half_t)a4[j].z, (half_t)a4[j].w};
          *(v4h*)(dst + j * 4) = h;
        }
      }
#pragma unroll
      for (int rep = 0; rep < 4; ++rep) {
        int k = bkg * 16 + rep * 4;
        v4h hl = {(half_t)bl[rep * 4], (half_t)bl[rep * 4 + 1],
                  (half_t)bl[rep * 4 + 2], (half_t)bl[rep * 4 + 3]};
        *(v4h*)&u.l1.B1lo[nb][bn][k] = hl;
        v4h hh = {(half_t)bh[rep * 4], (half_t)bh[rep * 4 + 1],
                  (half_t)bh[rep * 4 + 2], (half_t)bh[rep * 4 + 3]};
        *(v4h*)&u.l1.B1hi[nb][bn][k] = hh;
      }
    }
    __syncthreads();
  }

  auto tail = [&](const v8f (&acc)[8], int e, int invert, const float* rws) {
    __syncthreads();  // all layer1 LDS reads done before overlaying union
    const float* w2e = w2 + (size_t)e * H_ * H_;
    const float* w3e = w3 + (size_t)e * H_ * O_;
    if (tid < H_) { b1s[tid] = b1[e * H_ + tid]; b2s[tid] = b2[e * H_ + tid]; }
    if (tid < 16) b3s[tid] = (tid < O_) ? b3[e * O_ + tid] : 0.f;
    {  // stage w2 transposed: lane owns col bn, 4 consecutive k per packed store
#pragma unroll
      for (int rep = 0; rep < 16; ++rep) {
        int k = bkg * 64 + rep * 4;
        v4h h = {(half_t)w2e[(size_t)(k + 0) * H_ + bn], (half_t)w2e[(size_t)(k + 1) * H_ + bn],
                 (half_t)w2e[(size_t)(k + 2) * H_ + bn], (half_t)w2e[(size_t)(k + 3) * H_ + bn]};
        *(v4h*)&u.t.W2t[bn][k] = h;
      }
    }
    if (tid < H_) {  // stage w3 transposed, zero-pad O..15
#pragma unroll
      for (int o = 0; o < 16; ++o)
        u.t.W3t[o][tid] = (o < O_) ? (half_t)w3e[(size_t)tid * O_ + o] : (half_t)0.f;
    }
    __syncthreads();

    {  // h1 = relu(acc + b1) -> Hbuf (each wave writes its own 16 rows)
      int l = lane & 15, hi = lane >> 4;
#pragma unroll
      for (int ct = 0; ct < 8; ++ct)
#pragma unroll
        for (int i = 0; i < 8; ++i) {
          int rr = wid * 16 + (hi ? 8 + i : i);
          int col = ct * 16 + l;
          float vv = acc[ct][i] + b1s[col];
          Hbuf[rr][col] = (half_t)(vv > 0.f ? vv : 0.f);
        }
    }
    // layer2 (A rows == the rows this wave just wrote; no barrier needed)
    v8f a2[8];
#pragma unroll
    for (int i = 0; i < 8; ++i) vzero(a2[i]);
#pragma unroll
    for (int kc2 = 0; kc2 < H_; kc2 += 32) {
      v16h aa = ldsA(&Hbuf[0][0], 136, wid * 16, kc2, lane);
#pragma unroll
      for (int ct = 0; ct < 8; ++ct) {
        v16h bb2 = ldsB(&u.t.W2t[0][0], 136, ct * 16, kc2, lane);
        a2[ct] = wmma16(aa, bb2, a2[ct]);
      }
    }
    {  // h2 = relu(a2 + b2) -> H2buf
      int l = lane & 15, hi = lane >> 4;
#pragma unroll
      for (int ct = 0; ct < 8; ++ct)
#pragma unroll
        for (int i = 0; i < 8; ++i) {
          int rr = wid * 16 + (hi ? 8 + i : i);
          int col = ct * 16 + l;
          float vv = a2[ct][i] + b2s[col];
          H2buf[rr][col] = (half_t)(vv > 0.f ? vv : 0.f);
        }
    }
    // layer3: 16-wide output tile (cols 0..9 valid)
    v8f a3; vzero(a3);
#pragma unroll
    for (int kc2 = 0; kc2 < H_; kc2 += 32) {
      v16h aa = ldsA(&H2buf[0][0], 136, wid * 16, kc2, lane);
      v16h bb3 = ldsB(&u.t.W3t[0][0], 136, 0, kc2, lane);
      a3 = wmma16(aa, bb3, a3);
    }
    {  // weighted scatter to per-(row,slot) partials (deterministic, no atomics)
      int l = lane & 15, hi = lane >> 4;
      if (l < O_) {
#pragma unroll
        for (int i = 0; i < 8; ++i) {
          int rr = wid * 16 + (hi ? 8 + i : i);
          if (rr < nvalid) {
            int pk = rowp[rr];
            int rid = pk & 0x3FFF;
            int slo = (pk >> 14) & 1;
            int slot = invert ? (1 - slo) : slo;
            partial[((size_t)rid * 2 + slot) * O_ + l] = rws[rr] * (a3[i] + b3s[l]);
          }
        }
      }
    }
  };

  tail(accL, eLo, 0, rwl);
  tail(accH, eHi, 1, rwh);
}

// ---------------------------------------------------------------------------
// Kernel 3: combine the two expert slots per row
// ---------------------------------------------------------------------------
__global__ __launch_bounds__(256) void moe_combine(const float* __restrict__ partial,
                                                   float* __restrict__ y) {
  int i = blockIdx.x * 256 + threadIdx.x;
  if (i < B_ * O_) {
    int b = i / O_, o = i - b * O_;
    y[i] = partial[((size_t)b * 2) * O_ + o] + partial[((size_t)b * 2 + 1) * O_ + o];
  }
}

extern "C" void kernel_launch(void* const* d_in, const int* in_sizes, int n_in,
                              void* d_out, int out_size, void* d_ws, size_t ws_size,
                              hipStream_t stream) {
  (void)in_sizes; (void)n_in; (void)out_size; (void)ws_size;
  const float* x  = (const float*)d_in[0];
  const float* gw = (const float*)d_in[1];
  const float* gb = (const float*)d_in[2];
  const float* w1 = (const float*)d_in[3];
  const float* b1 = (const float*)d_in[4];
  const float* w2 = (const float*)d_in[5];
  const float* b2 = (const float*)d_in[6];
  const float* w3 = (const float*)d_in[7];
  const float* b3 = (const float*)d_in[8];

  char* ws = (char*)d_ws;
  int*   cnt     = (int*)ws;
  int*   bucket  = (int*)(ws + 128);
  float* wlo     = (float*)(ws + 128 + (size_t)NPAIR * B_ * 4);
  float* whi     = (float*)(ws + 128 + (size_t)NPAIR * B_ * 8);
  float* partial = (float*)(ws + 128 + (size_t)NPAIR * B_ * 12);

  moe_zero<<<1, 32, 0, stream>>>(cnt);
  moe_gate<<<B_ / MT, 256, 0, stream>>>(x, gw, gb, cnt, bucket, wlo, whi);
  moe_ffn<<<NPAIR * TILES, 256, 0, stream>>>(x, w1, b1, w2, b2, w3, b3,
                                             cnt, bucket, wlo, whi, partial);
  moe_combine<<<(B_ * O_ + 255) / 256, 256, 0, stream>>>(partial, (float*)d_out);
}
